// _SparseMLP_64518998721141
// MI455X (gfx1250) — matrix operand order repaired and checked by reading, not compiled, not run on hardware
//
#include <hip/hip_runtime.h>
#include <hip/hip_bf16.h>
#include <stdint.h>

// ---------------- problem constants ----------------
#define T_TOK 4096   // B*S
#define HDIM  2048
#define NEXP  8
#define IDIM  1408
#define SIDIM 2816   // I * num_shared_expert

// ---------------- GEMM tiling ----------------
#define BM 128
#define BN 64
#define KB 32
#define LDA 48   // LDS halves per A row (96B rows -> 16B-aligned b128 accesses)
#define LDB 40   // LDS halves per B^T row (80B rows, 8B-aligned b64 loads)

typedef __bf16 bf16_t;
typedef __attribute__((ext_vector_type(16))) __bf16 v16bf;
typedef __attribute__((ext_vector_type(8)))  float  v8f;

#define WMMA_BF16(A,B,C) \
  __builtin_amdgcn_wmma_f32_16x16x32_bf16(false,(A),false,(B),(short)0,(C),false,false)

static __device__ __forceinline__ v8f zerov() {
  v8f z;
#pragma unroll
  for (int i = 0; i < 8; ++i) z[i] = 0.f;
  return z;
}

// async copy 32 bytes global(bf16) -> LDS, tracked with ASYNCcnt
static __device__ __forceinline__ void async_copy32(bf16_t* lds, const bf16_t* g) {
  const uint32_t lo = (uint32_t)(uintptr_t)lds;   // low 32 bits = LDS byte offset
  asm volatile("global_load_async_to_lds_b128 %0, %1, off"
               :: "v"(lo), "v"(g) : "memory");
  asm volatile("global_load_async_to_lds_b128 %0, %1, off offset:16"
               :: "v"(lo), "v"(g) : "memory");
}
static __device__ __forceinline__ void wait_async0() {
  asm volatile("s_wait_asynccnt 0" ::: "memory");
}

// A fragment, 16x32 bf16: lanes 0-15 = rows, K = 16*(lane/16)+j
static __device__ __forceinline__ v16bf frag_a(const bf16_t* As, int mSub, int lane) {
  const int row = mSub + (lane & 15);
  const int hi  = (lane >> 4) << 4;   // 0 or 16
  union { v16bf v; uint4 q[2]; } u;
  u.q[0] = *(const uint4*)(As + row * LDA + hi);        // 16B aligned (96*row + 2*hi)
  u.q[1] = *(const uint4*)(As + row * LDA + 8 + hi);
  return u.v;
}

// B fragment, 32x16 bf16 from LDS stored transposed Bt[n][k]: lane%16 = col, K = 16*(lane/16)+j
static __device__ __forceinline__ v16bf frag_b(const bf16_t* Bt, int nSub, int lane) {
  const int col = nSub + (lane & 15);
  const int grp = (lane >> 4) << 4;   // 0 or 16
  union { v16bf v; uint2 q[4]; } u;
  const uint2* p = (const uint2*)(Bt + col * LDB + grp);
  u.q[0] = p[0]; u.q[1] = p[1]; u.q[2] = p[2]; u.q[3] = p[3];
  return u.v;
}

// ---------------- small kernels ----------------
__global__ void zero_counts_k(int* counts) {
  if (threadIdx.x < NEXP) counts[threadIdx.x] = 0;
}

// one wave32 per token: 8 logits, top-2, renormalized weights (softmax denom cancels)
__global__ __launch_bounds__(256) void router_k(const float* __restrict__ x,
                                                const float* __restrict__ rw,
                                                int* __restrict__ counts,
                                                int* __restrict__ tokIdx,
                                                float* __restrict__ tokW) {
  const int wid  = threadIdx.x >> 5;
  const int lane = threadIdx.x & 31;
  const int t    = blockIdx.x * 8 + wid;
  float acc[NEXP];
#pragma unroll
  for (int e = 0; e < NEXP; ++e) acc[e] = 0.f;
  const float* xr = x + (size_t)t * HDIM;
  for (int k = lane; k < HDIM; k += 32) {
    const float xv = xr[k];
#pragma unroll
    for (int e = 0; e < NEXP; ++e) acc[e] += xv * rw[e * HDIM + k];
  }
#pragma unroll
  for (int e = 0; e < NEXP; ++e)
#pragma unroll
    for (int off = 16; off > 0; off >>= 1) acc[e] += __shfl_xor(acc[e], off, 32);
  if (lane == 0) {
    int i1 = 0;
#pragma unroll
    for (int e = 1; e < NEXP; ++e) if (acc[e] > acc[i1]) i1 = e;
    int i2 = (i1 == 0) ? 1 : 0;
#pragma unroll
    for (int e = 0; e < NEXP; ++e) if (e != i1 && acc[e] > acc[i2]) i2 = e;
    const float w1 = 1.f / (1.f + __expf(acc[i2] - acc[i1]));  // p1/(p1+p2)
    const float w2 = 1.f - w1;
    int p1 = atomicAdd(&counts[i1], 1);
    tokIdx[i1 * T_TOK + p1] = t; tokW[i1 * T_TOK + p1] = w1;
    int p2 = atomicAdd(&counts[i2], 1);
    tokIdx[i2 * T_TOK + p2] = t; tokW[i2 * T_TOK + p2] = w2;
  }
}

__global__ void prefix_k(const int* __restrict__ counts, int* __restrict__ offsets) {
  if (threadIdx.x == 0) {
    int s = 0;
    for (int e = 0; e < NEXP; ++e) { offsets[e] = s; s += counts[e]; }
    offsets[NEXP] = s;
  }
}

__global__ void cast_x_k(const float* __restrict__ x, bf16_t* __restrict__ xbf) {
  const int i = (blockIdx.x * 256 + threadIdx.x) * 4;
  const float4 v = *(const float4*)(x + i);
  union { uint2 q; bf16_t b[4]; } u;
  u.b[0] = (bf16_t)v.x; u.b[1] = (bf16_t)v.y; u.b[2] = (bf16_t)v.z; u.b[3] = (bf16_t)v.w;
  *(uint2*)(xbf + i) = u.q;
}

// d_out[t*H + h] = s_down_b[h]; every accumulation afterwards is atomicAdd
__global__ void init_out_k(float* __restrict__ out, const float* __restrict__ bias) {
  const int i = blockIdx.x * 256 + threadIdx.x;
  out[i] = bias[i & (HDIM - 1)];
}

// ---------------- fused gate+up SwiGLU GEMM (async A staging + ping-pong LDS) -----------
template <bool ROUTED>
__global__ __launch_bounds__(256)
void gateup_gemm(const bf16_t* __restrict__ xbf,
                 const float* __restrict__ gwBase, const float* __restrict__ uwBase,
                 size_t wStride,
                 const float* __restrict__ gb, const float* __restrict__ ub,
                 bf16_t* __restrict__ hout,
                 int N, int K, int Mtot,
                 const int* __restrict__ counts, const int* __restrict__ offsets,
                 const int* __restrict__ tokIdx) {
  __shared__ bf16_t As[2][BM * LDA];
  __shared__ bf16_t Bg[2][BN * LDB];
  __shared__ bf16_t Bu[2][BN * LDB];
  __shared__ int    sTok[BM];

  const int e     = blockIdx.z;
  const int mBase = blockIdx.y * BM;
  const int nBase = blockIdx.x * BN;
  const int cnt   = ROUTED ? counts[e] : Mtot;
  if (mBase >= cnt) return;
  const int offs  = ROUTED ? offsets[e] : 0;
  const float* gw = gwBase + (size_t)e * wStride;
  const float* uw = uwBase + (size_t)e * wStride;

  const int t = threadIdx.x;
  if (ROUTED && t < BM) {
    const int row = mBase + t;
    sTok[t] = (row < cnt) ? tokIdx[e * T_TOK + row] : -1;
  }
  __syncthreads();  // sTok visible before first async issue

  const int lane = t & 31, wid = t >> 5;
  const int wm = (wid & 3) * 32, wn = (wid >> 2) * 32;

  const int rA  = t >> 1;         // A row this thread stages
  const int seg = (t & 1) * 16;   // half-offset within the 32-wide K chunk
  int tokRow;
  if (ROUTED) tokRow = sTok[rA];
  else        tokRow = mBase + rA;
  // invalid rows: skip copy; stale LDS garbage only affects guarded output rows
  const bool aValid = (tokRow >= 0);
  const bf16_t* aRowP = aValid ? (xbf + (size_t)tokRow * K + seg) : nullptr;

  v8f accG[2][2], accU[2][2];
#pragma unroll
  for (int m = 0; m < 2; ++m)
#pragma unroll
    for (int j = 0; j < 2; ++j) { accG[m][j] = zerov(); accU[m][j] = zerov(); }

  float pg[8], pu[8];

  // prologue: stage k=0
  if (aValid) async_copy32(&As[0][rA * LDA + seg], aRowP);
#pragma unroll
  for (int i = 0; i < 8; ++i) {
    const int idx = i * 256 + t;
    const int kk = idx >> 6, nn = idx & 63;
    const size_t gi = (size_t)kk * N + nBase + nn;
    pg[i] = gw[gi]; pu[i] = uw[gi];
  }

  int cur = 0;
  for (int kBase = 0; kBase < K; kBase += KB) {
    // finish staging current buffer: cvt+store B, wait for async A
#pragma unroll
    for (int i = 0; i < 8; ++i) {
      const int idx = i * 256 + t;
      const int kk = idx >> 6, nn = idx & 63;
      Bg[cur][nn * LDB + kk] = (bf16_t)pg[i];
      Bu[cur][nn * LDB + kk] = (bf16_t)pu[i];
    }
    wait_async0();
    __syncthreads();  // cur staged; everyone done computing on cur^1

    const int kn = kBase + KB;
    if (kn < K) {  // prefetch next tile while computing
      if (aValid) async_copy32(&As[cur ^ 1][rA * LDA + seg], aRowP + kn);
#pragma unroll
      for (int i = 0; i < 8; ++i) {
        const int idx = i * 256 + t;
        const int kk = idx >> 6, nn = idx & 63;
        const size_t gi = (size_t)(kn + kk) * N + nBase + nn;
        pg[i] = gw[gi]; pu[i] = uw[gi];
      }
    }

    // compute on cur: load all fragments, then burst 8 WMMAs
    const v16bf a0  = frag_a(As[cur], wm,      lane);
    const v16bf a1  = frag_a(As[cur], wm + 16, lane);
    const v16bf bg0 = frag_b(Bg[cur], wn,      lane);
    const v16bf bg1 = frag_b(Bg[cur], wn + 16, lane);
    const v16bf bu0 = frag_b(Bu[cur], wn,      lane);
    const v16bf bu1 = frag_b(Bu[cur], wn + 16, lane);
    accG[0][0] = WMMA_BF16(a0, bg0, accG[0][0]);
    accG[1][0] = WMMA_BF16(a1, bg0, accG[1][0]);
    accG[0][1] = WMMA_BF16(a0, bg1, accG[0][1]);
    accG[1][1] = WMMA_BF16(a1, bg1, accG[1][1]);
    accU[0][0] = WMMA_BF16(a0, bu0, accU[0][0]);
    accU[1][0] = WMMA_BF16(a1, bu0, accU[1][0]);
    accU[0][1] = WMMA_BF16(a0, bu1, accU[0][1]);
    accU[1][1] = WMMA_BF16(a1, bu1, accU[1][1]);
    cur ^= 1;
  }

  // epilogue: silu(gate+gb) * (up+ub) -> bf16 intermediate
  const int colIn = lane & 15;
  const int rgrp  = (lane >> 4) * 8;
#pragma unroll
  for (int m = 0; m < 2; ++m)
#pragma unroll
    for (int j = 0; j < 2; ++j) {
      const int nCol = nBase + wn + j * 16 + colIn;
      const float gbv = gb ? gb[nCol] : 0.f;
      const float ubv = ub ? ub[nCol] : 0.f;
#pragma unroll
      for (int rr = 0; rr < 8; ++rr) {
        const int rowT = wm + m * 16 + rgrp + rr;
        const int row  = mBase + rowT;
        if (row < cnt) {
          const float g  = accG[m][j][rr] + gbv;
          const float u  = accU[m][j][rr] + ubv;
          const float hv = (g / (1.f + __expf(-g))) * u;
          hout[(size_t)(offs + row) * N + nCol] = (bf16_t)hv;
        }
      }
    }
}

// ---------------- down-projection GEMM, atomic-accumulated into d_out ----------------
template <bool ROUTED>
__global__ __launch_bounds__(256)
void down_gemm(const bf16_t* __restrict__ hbuf,
               const float* __restrict__ dwBase, size_t wStride,
               float* __restrict__ out,
               int N, int K, int Mtot,
               const int* __restrict__ counts, const int* __restrict__ offsets,
               const int* __restrict__ tokIdx, const float* __restrict__ tokW) {
  __shared__ bf16_t As[2][BM * LDA];
  __shared__ bf16_t Bd[2][BN * LDB];
  __shared__ int    sTok[BM];
  __shared__ float  sWt[BM];

  const int e     = blockIdx.z;
  const int mBase = blockIdx.y * BM;
  const int nBase = blockIdx.x * BN;
  const int cnt   = ROUTED ? counts[e] : Mtot;
  if (mBase >= cnt) return;
  const int offs  = ROUTED ? offsets[e] : 0;
  const float* dw = dwBase + (size_t)e * wStride;

  const int t = threadIdx.x;
  if (ROUTED && t < BM) {
    const int row = mBase + t;
    if (row < cnt) { sTok[t] = tokIdx[e * T_TOK + row]; sWt[t] = tokW[e * T_TOK + row]; }
    else           { sTok[t] = 0;                        sWt[t] = 0.f; }
  }
  __syncthreads();

  const int lane = t & 31, wid = t >> 5;
  const int wm = (wid & 3) * 32, wn = (wid >> 2) * 32;
  v8f acc[2][2];
#pragma unroll
  for (int m = 0; m < 2; ++m)
#pragma unroll
    for (int j = 0; j < 2; ++j) acc[m][j] = zerov();

  const int rA  = t >> 1;
  const int seg = (t & 1) * 16;
  const int row0 = mBase + rA;
  const bool aValid = (row0 < cnt);
  const bf16_t* aRowP = aValid ? (hbuf + (size_t)(offs + row0) * K + seg) : nullptr;

  float pd[8];

  if (aValid) async_copy32(&As[0][rA * LDA + seg], aRowP);
#pragma unroll
  for (int i = 0; i < 8; ++i) {
    const int idx = i * 256 + t;
    const int kk = idx >> 6, nn = idx & 63;
    pd[i] = dw[(size_t)kk * N + nBase + nn];
  }

  int cur = 0;
  for (int kBase = 0; kBase < K; kBase += KB) {
#pragma unroll
    for (int i = 0; i < 8; ++i) {
      const int idx = i * 256 + t;
      const int kk = idx >> 6, nn = idx & 63;
      Bd[cur][nn * LDB + kk] = (bf16_t)pd[i];
    }
    wait_async0();
    __syncthreads();

    const int kn = kBase + KB;
    if (kn < K) {
      if (aValid) async_copy32(&As[cur ^ 1][rA * LDA + seg], aRowP + kn);
#pragma unroll
      for (int i = 0; i < 8; ++i) {
        const int idx = i * 256 + t;
        const int kk = idx >> 6, nn = idx & 63;
        pd[i] = dw[(size_t)(kn + kk) * N + nBase + nn];
      }
    }

    const v16bf a0 = frag_a(As[cur], wm,      lane);
    const v16bf a1 = frag_a(As[cur], wm + 16, lane);
    const v16bf b0 = frag_b(Bd[cur], wn,      lane);
    const v16bf b1 = frag_b(Bd[cur], wn + 16, lane);
    acc[0][0] = WMMA_BF16(a0, b0, acc[0][0]);
    acc[1][0] = WMMA_BF16(a1, b0, acc[1][0]);
    acc[0][1] = WMMA_BF16(a0, b1, acc[0][1]);
    acc[1][1] = WMMA_BF16(a1, b1, acc[1][1]);
    cur ^= 1;
  }

  const int colIn = lane & 15;
  const int rgrp  = (lane >> 4) * 8;
#pragma unroll
  for (int m = 0; m < 2; ++m)
#pragma unroll
    for (int j = 0; j < 2; ++j) {
      const int nCol = nBase + wn + j * 16 + colIn;
#pragma unroll
      for (int rr = 0; rr < 8; ++rr) {
        const int rowT = wm + m * 16 + rgrp + rr;
        const int row  = mBase + rowT;
        if (row < cnt) {
          float v = acc[m][j][rr];
          int   tok;
          if (ROUTED) { tok = sTok[rowT]; v *= sWt[rowT]; }
          else        { tok = row; }
          unsafeAtomicAdd(&out[(size_t)tok * HDIM + nCol], v);
        }
      }
    }
}

// ---------------- host launcher ----------------
extern "C" void kernel_launch(void* const* d_in, const int* in_sizes, int n_in,
                              void* d_out, int out_size, void* d_ws, size_t ws_size,
                              hipStream_t stream) {
  const float* x   = (const float*)d_in[0];
  const float* rw  = (const float*)d_in[1];
  const float* gw  = (const float*)d_in[2];
  const float* uw  = (const float*)d_in[3];
  const float* dw  = (const float*)d_in[4];
  const float* sgw = (const float*)d_in[5];
  const float* sgb = (const float*)d_in[6];
  const float* suw = (const float*)d_in[7];
  const float* sub = (const float*)d_in[8];
  const float* sdw = (const float*)d_in[9];
  const float* sdb = (const float*)d_in[10];
  float* out = (float*)d_out;

  // workspace layout (~63 MB)
  char* ws = (char*)d_ws;
  bf16_t* xbf = (bf16_t*)ws; ws += (size_t)T_TOK * HDIM * 2;        // 16.8 MB
  bf16_t* hs  = (bf16_t*)ws; ws += (size_t)T_TOK * SIDIM * 2;       // 23.1 MB
  bf16_t* hx  = (bf16_t*)ws; ws += (size_t)2 * T_TOK * IDIM * 2;    // 23.1 MB (2T rows)
  int*    counts  = (int*)ws;   ws += 256;
  int*    offsets = (int*)ws;   ws += 256;
  int*    tokIdx  = (int*)ws;   ws += (size_t)NEXP * T_TOK * 4;     // 128 KB
  float*  tokW    = (float*)ws; ws += (size_t)NEXP * T_TOK * 4;     // 128 KB

  zero_counts_k<<<1, 32, 0, stream>>>(counts);
  router_k<<<T_TOK / 8, 256, 0, stream>>>(x, rw, counts, tokIdx, tokW);
  prefix_k<<<1, 1, 0, stream>>>(counts, offsets);
  cast_x_k<<<(T_TOK * HDIM) / (256 * 4), 256, 0, stream>>>(x, xbf);
  init_out_k<<<(T_TOK * HDIM) / 256, 256, 0, stream>>>(out, sdb);

  // shared expert: gate/up fused, then down (bias already in d_out)
  gateup_gemm<false><<<dim3(SIDIM / BN, T_TOK / BM, 1), 256, 0, stream>>>(
      xbf, sgw, suw, 0, sgb, sub, hs, SIDIM, HDIM, T_TOK, nullptr, nullptr, nullptr);
  down_gemm<false><<<dim3(HDIM / BN, T_TOK / BM, 1), 256, 0, stream>>>(
      hs, sdw, 0, out, HDIM, SIDIM, T_TOK, nullptr, nullptr, nullptr, nullptr);

  // routed experts: token-gathered gate/up fused, then scaled down accumulate
  gateup_gemm<true><<<dim3(IDIM / BN, T_TOK / BM, NEXP), 256, 0, stream>>>(
      xbf, gw, uw, (size_t)HDIM * IDIM, nullptr, nullptr, hx, IDIM, HDIM, T_TOK,
      counts, offsets, tokIdx);
  down_gemm<true><<<dim3(HDIM / BN, T_TOK / BM, NEXP), 256, 0, stream>>>(
      hx, dw, (size_t)IDIM * HDIM, out, HDIM, IDIM, T_TOK,
      counts, offsets, tokIdx, tokW);
}
